// SNNClassifier_61993557950646
// MI455X (gfx1250) — matrix operand order repaired and checked by reading, not compiled, not run on hardware
//
#include <hip/hip_runtime.h>
#include <stdint.h>

typedef __attribute__((ext_vector_type(16))) _Float16 v16h;
typedef __attribute__((ext_vector_type(8)))  float    v8f;

#define T_STEPS 500
#define BATCH   256
#define NIN     96
#define NHID    64
#define NOUT    80
#define BETA    0.95f
#define THETA   1.0f

// x tile: 16 rows x 96 f32 = 6144 B = 384 x 16B chunks, contiguous in global mem.
#define XTILE_FLOATS (16 * NIN)
#define XTILE_CHUNKS (XTILE_FLOATS / 4)

// Async DMA one x tile (global, contiguous) into LDS at byte offset lds_base.
// CDNA5 GLOBAL_LOAD_ASYNC_TO_LDS_B128: VDST = LDS byte address VGPR, VADDR = 64-bit
// global address VGPR pair; tracked by ASYNCcnt.
__device__ __forceinline__ void async_copy_xtile(const float* __restrict__ xt,
                                                 uint32_t lds_base, int tid) {
    for (int i = tid; i < XTILE_CHUNKS; i += 160) {
        uint32_t ldso = lds_base + 16u * (uint32_t)i;
        uint64_t gaddr = (uint64_t)(uintptr_t)(xt + 4 * i);
        asm volatile("global_load_async_to_lds_b128 %0, %1, off"
                     :: "v"(ldso), "v"(gaddr)
                     : "memory");
    }
}

__device__ __forceinline__ void wait_asynccnt0() {
    asm volatile("s_wait_asynccnt 0" ::: "memory");
}

// One block = 5 waves (wave32) = 160 threads, owns a 16-row batch tile for all T.
// waves 0..3: hidden-layer N-tiles (4 x 16 = 64); waves 0..4: output N-tiles (5 x 16 = 80).
__launch_bounds__(160, 1)
__global__ void snn_leaky2_kernel(const float* __restrict__ x,
                                  const float* __restrict__ W1,
                                  const float* __restrict__ b1,
                                  const float* __restrict__ W2,
                                  const float* __restrict__ b2,
                                  float* __restrict__ out)
{
    __shared__ float    lds_xf[2][XTILE_FLOATS]; // double-buffered x_t tile (f32, DMA'd)
    __shared__ _Float16 lds_s[16 * NHID];        // spk1 tile, row-major [m][h], f16

    const int tid    = threadIdx.x;
    const int wave   = tid >> 5;        // 0..4
    const int lane   = tid & 31;
    const int lo16   = lane & 15;
    const int hiHalf = lane >> 4;       // 0 or 1
    const int rowbase = blockIdx.x * 16;

    // ---- Preload weight B-fragments (constant across all 500 steps) ----
    // B 32x16 f16, contiguous k order chosen here (matches A): lanes 0-15 hold K=0..15 for col=lane&15, lanes 16-31 K=16..31.
    v16h w1f[3];
    if (wave < 4) {
        for (int c = 0; c < 3; ++c)
            for (int e = 0; e < 16; ++e) {
                int k = 32 * c + hiHalf * 16 + e;            // 0..95
                w1f[c][e] = (_Float16)W1[k * NHID + (wave * 16 + lo16)];
            }
    }
    v16h w2f[2];
    for (int c = 0; c < 2; ++c)
        for (int e = 0; e < 16; ++e) {
            int k = 32 * c + hiHalf * 16 + e;                // 0..63
            w2f[c][e] = (_Float16)W2[k * NOUT + (wave * 16 + lo16)];
        }
    const float bias1 = (wave < 4) ? b1[wave * 16 + lo16] : 0.0f;
    const float bias2 = b2[wave * 16 + lo16];

    // Membrane/spike state in WMMA C-fragment layout:
    //   vgpr i, lane L -> row m = i + 8*(L>=16), col n = L&15
    v8f mem1 = {}; v8f spk1p = {};
    v8f mem2 = {}; v8f spk2p = {};

    float* spk_out = out;
    float* mem_out = out + (size_t)T_STEPS * BATCH * NOUT;

    const uint32_t ldsx_base0 = (uint32_t)(uintptr_t)&lds_xf[0][0];
    const uint32_t ldsx_stride = XTILE_FLOATS * sizeof(float);

    // Prologue: DMA tile for t=0 into buffer 0.
    async_copy_xtile(x + (size_t)rowbase * NIN, ldsx_base0, tid);

#pragma unroll 1
    for (int t = 0; t < T_STEPS; ++t) {
        // Tile for step t was DMA'd one step ahead; wait + publish.
        wait_asynccnt0();
        __syncthreads();

        // Kick off DMA for step t+1 into the other buffer (its previous readers
        // all finished before the barrier above).
        if (t + 1 < T_STEPS)
            async_copy_xtile(x + ((size_t)(t + 1) * BATCH + rowbase) * NIN,
                             ldsx_base0 + ((t + 1) & 1) * ldsx_stride, tid);

        const float* xbuf = &lds_xf[t & 1][0];

        // ---- Layer 1: cur1 = x_t @ W1; LIF update; spikes -> LDS ----
        if (wave < 4) {
            v8f acc = {};
            for (int c = 0; c < 3; ++c) {
                // A 16x32 f16, contiguous k chosen here (matches B): lane L<16 row L holds K {0..15},
                //                     lane L>=16 row L-16 holds K {16..31}
                v16h a;
                for (int e = 0; e < 16; ++e) {
                    int k = 32 * c + hiHalf * 16 + e;
                    a[e] = (_Float16)xbuf[lo16 * NIN + k];
                }
                acc = __builtin_amdgcn_wmma_f32_16x16x32_f16(
                    false, a, false, w1f[c], (short)0, acc, false, false);
            }
            for (int i = 0; i < 8; ++i) {
                float m = BETA * mem1[i] + (acc[i] + bias1) - spk1p[i] * THETA;
                mem1[i] = m;
                float s = (m > THETA) ? 1.0f : 0.0f;
                spk1p[i] = s;
                int mrow = i + hiHalf * 8;
                lds_s[mrow * NHID + wave * 16 + lo16] = (_Float16)s;
            }
        }
        __syncthreads();

        // ---- Layer 2: cur2 = spk1 @ W2; LIF update; write spk2/mem2 ----
        v8f acc2 = {};
        for (int c = 0; c < 2; ++c) {
            v16h a;
            for (int e = 0; e < 16; ++e) {
                int k = 32 * c + hiHalf * 16 + e;
                a[e] = lds_s[lo16 * NHID + k];
            }
            acc2 = __builtin_amdgcn_wmma_f32_16x16x32_f16(
                false, a, false, w2f[c], (short)0, acc2, false, false);
        }
        const size_t obase =
            ((size_t)t * BATCH + rowbase) * NOUT + wave * 16 + lo16;
        for (int i = 0; i < 8; ++i) {
            float m = BETA * mem2[i] + (acc2[i] + bias2) - spk2p[i] * THETA;
            mem2[i] = m;
            float s = (m > THETA) ? 1.0f : 0.0f;
            spk2p[i] = s;
            int mrow = i + hiHalf * 8;
            size_t idx = obase + (size_t)mrow * NOUT;
            spk_out[idx] = s;   // coalesced along lanes (n contiguous)
            mem_out[idx] = m;
        }
        // No trailing barrier needed: next step's top wait+barrier protects
        // lds_s and the x buffer being overwritten.
    }
}

extern "C" void kernel_launch(void* const* d_in, const int* in_sizes, int n_inputs,
                              void* d_out, int out_size, void* d_ws, size_t ws_size,
                              hipStream_t stream) {
    const float* x  = (const float*)d_in[0];   // (500, 256, 96)
    const float* W1 = (const float*)d_in[1];   // (96, 64)
    const float* b1 = (const float*)d_in[2];   // (64,)
    const float* W2 = (const float*)d_in[3];   // (64, 80)
    const float* b2 = (const float*)d_in[4];   // (80,)
    float* out = (float*)d_out;                // spk(500,256,80) ++ mem(500,256,80)

    snn_leaky2_kernel<<<dim3(BATCH / 16), dim3(160), 0, stream>>>(
        x, W1, b1, W2, b2, out);
}
